// RNN_90958817394975
// MI455X (gfx1250) — matrix operand order repaired and checked by reading, not compiled, not run on hardware
//
#include <hip/hip_runtime.h>
#include <hip/hip_bf16.h>

// ---------------------------------------------------------------------------
// Problem constants
// ---------------------------------------------------------------------------
#define BB   32
#define SS   128
#define TT   127          // S-1 timesteps
#define VV   32000
#define EE   300
#define HH   1024
#define G3   3072         // 3*H
#define MM   4064         // TT*BB rows of hidden history

typedef __attribute__((ext_vector_type(16))) __bf16 v16bf;
typedef __attribute__((ext_vector_type(8)))  __bf16 v8bf;
typedef __attribute__((ext_vector_type(8)))  float  v8f;

union FragBF { v16bf v; v8bf h[2]; };

__device__ __forceinline__ __bf16 f2bf(float f) {
  union { float f; unsigned u; } x; x.f = f;
  unsigned r = x.u + 0x7FFFu + ((x.u >> 16) & 1u);   // round-to-nearest-even
  unsigned short hs = (unsigned short)(r >> 16);
  __bf16 out;
  __builtin_memcpy(&out, &hs, 2);
  return out;
}

// A fragment: 16x32 bf16, A row-major with leading dim lda.
// lane l16 -> row M = mb + l16 ; half selects K sub-ranges per ISA layout.
__device__ __forceinline__ v16bf load_a(const __bf16* __restrict__ A, int lda,
                                        int mb, int kb, int l16, int half) {
  const __bf16* p = A + (long long)(mb + l16) * lda + kb + half * 8;
  FragBF f;
  f.h[0] = *reinterpret_cast<const v8bf*>(p);        // K = kb + half*8  .. +7
  f.h[1] = *reinterpret_cast<const v8bf*>(p + 16);   // K = kb+16+half*8 .. +7
  return f.v;
}

// B fragment: 32x16 bf16 of W^T where W is (N x K) row-major, ldw = K.
// lane l16 -> column N = nb + l16 ; half selects K sub-ranges per ISA layout.
__device__ __forceinline__ v16bf load_b(const __bf16* __restrict__ W, int ldw,
                                        int nb, int kb, int l16, int half) {
  const __bf16* p = W + (long long)(nb + l16) * ldw + kb + half * 8;
  FragBF f;
  f.h[0] = *reinterpret_cast<const v8bf*>(p);        // K = kb + half*8  .. +7
  f.h[1] = *reinterpret_cast<const v8bf*>(p + 16);   // K = kb+16+half*8 .. +7
  return f.v;
}

__device__ __forceinline__ v8f wmma_bf16(v16bf a, v16bf b, v8f c) {
  return __builtin_amdgcn_wmma_f32_16x16x32_bf16(false, a, false, b,
                                                 (short)0, c, false, false);
}

__device__ __forceinline__ float sigmoidf(float x) {
  return 1.0f / (1.0f + expf(-x));
}

// ---------------------------------------------------------------------------
// Utility kernels
// ---------------------------------------------------------------------------
__global__ void cvt_bf16_kernel(const float* __restrict__ s,
                                __bf16* __restrict__ d, long long n) {
  long long i = (long long)blockIdx.x * blockDim.x + threadIdx.x;
  long long st = (long long)gridDim.x * blockDim.x;
  for (; i < n; i += st) d[i] = f2bf(s[i]);
}

__global__ void zero_u32_kernel(unsigned* __restrict__ p, long long n) {
  long long i = (long long)blockIdx.x * blockDim.x + threadIdx.x;
  long long st = (long long)gridDim.x * blockDim.x;
  for (; i < n; i += st) p[i] = 0u;
}

// gi0[row, col] = b_ih0[col] + emb[sent[b,t], :] . W_ih0[col, :]   (K = 300)
__global__ void gi0_kernel(const int* __restrict__ sent,
                           const float* __restrict__ emb,
                           const float* __restrict__ Wih0,
                           const float* __restrict__ bih0,
                           float* __restrict__ gi0) {
  int col = blockIdx.x * 256 + threadIdx.x;   // 0..3071
  int row = blockIdx.y;                       // 0..4063  (row = t*32 + b)
  int t = row >> 5, b = row & 31;
  int tok = sent[b * SS + t];
  const float4* x = reinterpret_cast<const float4*>(emb + (long long)tok * EE);
  const float4* w = reinterpret_cast<const float4*>(Wih0 + (long long)col * EE);
  float acc = bih0[col];
#pragma unroll 5
  for (int i = 0; i < EE / 4; ++i) {
    float4 xv = x[i], wv = w[i];
    acc += xv.x * wv.x + xv.y * wv.y + xv.z * wv.z + xv.w * wv.w;
  }
  gi0[(long long)row * G3 + col] = acc;
}

// ---------------------------------------------------------------------------
// One GRU layer step: gh = h_prev @ W_ghᵀ (+gi via GEMM or precomputed), gates.
// Grid: 32 blocks x 128 threads = 128 waves; wave -> (jt 0..63, mt 0..1).
// ---------------------------------------------------------------------------
template <bool GI_GEMM>
__global__ __launch_bounds__(128)
void gru_step_kernel(const __bf16* __restrict__ Agh,   // h_prev bf16 (32xH)
                     const __bf16* __restrict__ Wgh,   // (3H x H) bf16
                     const float*  __restrict__ bgh,   // 3H
                     const __bf16* __restrict__ Agi,   // layer1: x (=h0_new)
                     const __bf16* __restrict__ Wgi,   // layer1: W_ih bf16
                     const float*  __restrict__ bgi,   // layer1: b_ih
                     const float*  __restrict__ gi_pre,// layer0: gi0 row block
                     float*  __restrict__ hf,          // fp32 h, updated inplace
                     __bf16* __restrict__ hout) {      // new h bf16 (32xH)
  const int lane = threadIdx.x & 31, l16 = lane & 15, half = lane >> 4;
  const int wid = blockIdx.x * (blockDim.x >> 5) + (threadIdx.x >> 5);
  const int mt = wid & 1, jt = wid >> 1;
  const int mb = mt * 16, jb = jt * 16;

  v8f agh[3] = {};
  v8f agi[3] = {};
  for (int kb = 0; kb < HH; kb += 32) {
    v16bf a = load_a(Agh, HH, mb, kb, l16, half);
#pragma unroll
    for (int g = 0; g < 3; ++g) {
      v16bf b = load_b(Wgh, HH, g * HH + jb, kb, l16, half);
      agh[g] = wmma_bf16(a, b, agh[g]);
    }
    if constexpr (GI_GEMM) {
      v16bf a2 = load_a(Agi, HH, mb, kb, l16, half);
#pragma unroll
      for (int g = 0; g < 3; ++g) {
        v16bf b2 = load_b(Wgi, HH, g * HH + jb, kb, l16, half);
        agi[g] = wmma_bf16(a2, b2, agi[g]);
      }
    }
  }

#pragma unroll
  for (int i = 0; i < 8; ++i) {
    int m = mb + i + 8 * half;       // batch row 0..31
    int j = jb + l16;                // hidden index 0..1023
    float ghr = agh[0][i] + bgh[j];
    float ghz = agh[1][i] + bgh[HH + j];
    float ghn = agh[2][i] + bgh[2 * HH + j];
    float gir, giz, gin;
    if constexpr (GI_GEMM) {
      gir = agi[0][i] + bgi[j];
      giz = agi[1][i] + bgi[HH + j];
      gin = agi[2][i] + bgi[2 * HH + j];
    } else {
      const float* gp = gi_pre + (long long)m * G3;
      gir = gp[j]; giz = gp[HH + j]; gin = gp[2 * HH + j];
    }
    float r = sigmoidf(gir + ghr);
    float z = sigmoidf(giz + ghz);
    float n = tanhf(gin + r * ghn);
    float hp = hf[m * HH + j];
    float hn = (1.0f - z) * n + z * hp;
    hf[m * HH + j] = hn;
    hout[m * HH + j] = f2bf(hn);
  }
}

// ---------------------------------------------------------------------------
// Output projection: logits = Hall(4064xH) @ W_outᵀ(VxH) + b_out
// Written to d_out[1 + (b*127 + t)*V + v].
// Block: 128 threads (4 waves), each wave a 32x64 macro-tile (8 accums).
// A-tile (32x32 bf16, shared by all 4 waves) is double-buffered in LDS with
// an 80B row pitch so the 16-lane ds b128 reads hit 16 distinct banks.
// Grid: (V/256, MM/32).
// ---------------------------------------------------------------------------
#define APITCH 40   // bf16 elements per LDS row (80 bytes)

__global__ __launch_bounds__(128)
void logits_kernel(const __bf16* __restrict__ Hall,
                   const __bf16* __restrict__ Wout,
                   const float*  __restrict__ bout,
                   float* __restrict__ out) {
  __shared__ __bf16 asmem[2][32 * APITCH];
  const int tid = threadIdx.x;
  const int lane = tid & 31, l16 = lane & 15, half = lane >> 4;
  const int w = tid >> 5;                        // 0..3
  const int mb = blockIdx.y * 32;
  const int n0 = blockIdx.x * 256 + w * 64;      // wave covers 64 columns

  // Cooperative stage of the 32x32 A tile: 128 threads x 16B.
  auto stage = [&](int buf, int kb) {
    int row = tid >> 2, seg = tid & 3;           // 32 rows x 4 segments
    const v8bf* src = reinterpret_cast<const v8bf*>(
        Hall + (long long)(mb + row) * HH + kb + seg * 8);
    *reinterpret_cast<v8bf*>(&asmem[buf][row * APITCH + seg * 8]) = *src;
  };

  v8f acc[2][4] = {};
  stage(0, 0);
  __syncthreads();

  for (int kb = 0; kb < HH; kb += 32) {
    const int cur = (kb >> 5) & 1;
    if (kb + 32 < HH) stage(cur ^ 1, kb + 32);   // overlap next-tile staging

    // Prefetch the B stream two K-steps ahead (global_prefetch_b8).
    if (kb + 64 < HH)
      __builtin_prefetch(Wout + (long long)(n0 + l16) * HH + kb + 64, 0, 1);

    // A fragments from LDS (row-major, same per-lane layout as global).
    v16bf a[2];
#pragma unroll
    for (int mi = 0; mi < 2; ++mi) {
      const __bf16* p = &asmem[cur][(mi * 16 + l16) * APITCH + half * 8];
      FragBF f;
      f.h[0] = *reinterpret_cast<const v8bf*>(p);
      f.h[1] = *reinterpret_cast<const v8bf*>(p + 16);
      a[mi] = f.v;
    }

#pragma unroll
    for (int ni = 0; ni < 4; ++ni) {
      v16bf b = load_b(Wout, HH, n0 + ni * 16, kb, l16, half);
      acc[0][ni] = wmma_bf16(a[0], b, acc[0][ni]);
      acc[1][ni] = wmma_bf16(a[1], b, acc[1][ni]);
    }
    __syncthreads();   // staging of next buffer + LDS reads of current done
  }

#pragma unroll
  for (int mi = 0; mi < 2; ++mi) {
#pragma unroll
    for (int ni = 0; ni < 4; ++ni) {
#pragma unroll
      for (int i = 0; i < 8; ++i) {
        int m = mb + mi * 16 + i + 8 * half;   // row = t*32 + b
        int t = m >> 5, b = m & 31;
        int v = n0 + ni * 16 + l16;
        out[(long long)(b * TT + t) * VV + v] = acc[mi][ni][i] + bout[v];
      }
    }
  }
}

// ---------------------------------------------------------------------------
// Per-(b,t) log-sum-exp over V and gather at target token.
// ---------------------------------------------------------------------------
__global__ __launch_bounds__(256)
void lse_gather_kernel(const float* __restrict__ logits, // d_out+1, (b*TT+t)*V
                       const int* __restrict__ sent,
                       float* __restrict__ gathered) {
  int r = blockIdx.x;                 // 0..4063, r = b*TT + t
  int b = r / TT, t = r - b * TT;
  const float* row = logits + (long long)r * VV;
  __shared__ float sm[256];

  float mx = -3.4e38f;
  for (int i = threadIdx.x; i < VV; i += 256) mx = fmaxf(mx, row[i]);
  sm[threadIdx.x] = mx; __syncthreads();
  for (int s = 128; s > 0; s >>= 1) {
    if (threadIdx.x < s) sm[threadIdx.x] = fmaxf(sm[threadIdx.x], sm[threadIdx.x + s]);
    __syncthreads();
  }
  mx = sm[0]; __syncthreads();

  float sum = 0.0f;
  for (int i = threadIdx.x; i < VV; i += 256) sum += expf(row[i] - mx);
  sm[threadIdx.x] = sum; __syncthreads();
  for (int s = 128; s > 0; s >>= 1) {
    if (threadIdx.x < s) sm[threadIdx.x] += sm[threadIdx.x + s];
    __syncthreads();
  }
  if (threadIdx.x == 0) {
    int tok = sent[b * SS + t + 1];
    gathered[r] = row[tok] - mx - logf(sm[0]);
  }
}

__global__ __launch_bounds__(256)
void loss_kernel(const float* __restrict__ gathered,
                 const int* __restrict__ length,
                 float* __restrict__ dout) {
  __shared__ float sm[256];
  float acc = 0.0f;
  for (int idx = threadIdx.x; idx < BB * TT; idx += 256) {
    int b = idx / TT, t = idx - b * TT;
    int L = length[b] - 1;
    if (t < L) acc += gathered[idx] / (float)L;
  }
  sm[threadIdx.x] = acc; __syncthreads();
  for (int s = 128; s > 0; s >>= 1) {
    if (threadIdx.x < s) sm[threadIdx.x] += sm[threadIdx.x + s];
    __syncthreads();
  }
  if (threadIdx.x == 0) dout[0] = -sm[0] / (float)BB;
}

// ---------------------------------------------------------------------------
// Host side
// ---------------------------------------------------------------------------
extern "C" void kernel_launch(void* const* d_in, const int* in_sizes, int n_in,
                              void* d_out, int out_size, void* d_ws, size_t ws_size,
                              hipStream_t stream) {
  (void)in_sizes; (void)n_in; (void)out_size; (void)ws_size;

  const int*   sent  = (const int*)  d_in[0];
  const int*   length= (const int*)  d_in[1];
  const float* emb   = (const float*)d_in[2];
  const float* Wih0  = (const float*)d_in[3];
  const float* Whh0  = (const float*)d_in[4];
  const float* bih0  = (const float*)d_in[5];
  const float* bhh0  = (const float*)d_in[6];
  const float* Wih1  = (const float*)d_in[7];
  const float* Whh1  = (const float*)d_in[8];
  const float* bih1  = (const float*)d_in[9];
  const float* bhh1  = (const float*)d_in[10];
  const float* Wout  = (const float*)d_in[11];
  const float* bout  = (const float*)d_in[12];

  float* out = (float*)d_out;               // [loss, logits(b,t,v)]

  // Workspace carve-up (256B aligned).
  char* ws = (char*)d_ws;
  size_t off = 0;
  auto carve = [&](size_t bytes) {
    size_t r = off;
    off = (off + bytes + 255) & ~(size_t)255;
    return r;
  };
  __bf16* Wout_bf = (__bf16*)(ws + carve((size_t)VV * HH * 2));
  __bf16* Whh0_bf = (__bf16*)(ws + carve((size_t)G3 * HH * 2));
  __bf16* Wih1_bf = (__bf16*)(ws + carve((size_t)G3 * HH * 2));
  __bf16* Whh1_bf = (__bf16*)(ws + carve((size_t)G3 * HH * 2));
  __bf16* Hall_bf = (__bf16*)(ws + carve((size_t)MM * HH * 2));
  size_t zbase = off;                                  // zero region start
  __bf16* h1zero  = (__bf16*)(ws + carve((size_t)BB * HH * 2));
  __bf16* h0A     = (__bf16*)(ws + carve((size_t)BB * HH * 2));
  float*  h0f     = (float*) (ws + carve((size_t)BB * HH * 4));
  float*  h1f     = (float*) (ws + carve((size_t)BB * HH * 4));
  size_t zend = off;                                   // zero region end
  __bf16* h0B     = (__bf16*)(ws + carve((size_t)BB * HH * 2));
  float*  gi0     = (float*) (ws + carve((size_t)MM * G3 * 4));
  float*  gath    = (float*) (ws + carve((size_t)MM * 4));

  // 1) fp32 -> bf16 weight conversion.
  cvt_bf16_kernel<<<4096, 256, 0, stream>>>(Wout, Wout_bf, (long long)VV * HH);
  cvt_bf16_kernel<<<1024, 256, 0, stream>>>(Whh0, Whh0_bf, (long long)G3 * HH);
  cvt_bf16_kernel<<<1024, 256, 0, stream>>>(Wih1, Wih1_bf, (long long)G3 * HH);
  cvt_bf16_kernel<<<1024, 256, 0, stream>>>(Whh1, Whh1_bf, (long long)G3 * HH);

  // 2) Zero-init h state (h1zero, h0A, h0f, h1f are contiguous).
  zero_u32_kernel<<<128, 256, 0, stream>>>((unsigned*)(ws + zbase),
                                           (long long)((zend - zbase) / 4));

  // 3) Precompute all layer-0 input gates (parallel over all timesteps).
  gi0_kernel<<<dim3(G3 / 256, MM), 256, 0, stream>>>(sent, emb, Wih0, bih0, gi0);

  // 4) Sequential GRU recurrence: 127 x (layer0, layer1).
  for (int t = 0; t < TT; ++t) {
    const __bf16* h0_old = (t & 1) ? h0B : h0A;
    __bf16*       h0_new = (t & 1) ? h0A : h0B;
    gru_step_kernel<false><<<32, 128, 0, stream>>>(
        h0_old, Whh0_bf, bhh0,
        (const __bf16*)nullptr, (const __bf16*)nullptr, (const float*)nullptr,
        gi0 + (long long)t * BB * G3, h0f, h0_new);

    const __bf16* h1_old = (t == 0) ? h1zero : Hall_bf + (long long)(t - 1) * BB * HH;
    __bf16*       h1_new = Hall_bf + (long long)t * BB * HH;
    gru_step_kernel<true><<<32, 128, 0, stream>>>(
        h1_old, Whh1_bf, bhh1,
        h0_new, Wih1_bf, bih1,
        (const float*)nullptr, h1f, h1_new);
  }

  // 5) One big WMMA GEMM for all logits (266 GFLOP, W_out lives in L2).
  logits_kernel<<<dim3(VV / 256, MM / 32), 128, 0, stream>>>(
      Hall_bf, Wout_bf, bout, out + 1);

  // 6) log_softmax gather + masked length-normalized loss.
  lse_gather_kernel<<<MM, 256, 0, stream>>>(out + 1, sent, gath);
  loss_kernel<<<1, 256, 0, stream>>>(gath, length, out);
}
